// RepFlowLayerV7_22986664968683
// MI455X (gfx1250) — matrix operand order repaired and checked by reading, not compiled, not run on hardware
//
#include <hip/hip_runtime.h>
#include <hip/hip_bf16.h>

// ---------------------------------------------------------------------------
// RepFlow layer for gfx1250: all matmuls via v_wmma_f32_16x16x32_bf16.
// Activations converted f32->bf16 into LDS A-tiles; weights pre-packed once per
// launch into bf16 B-fragment order (32B contiguous per lane per k-tile).
// Segment softmax over 8 contiguous angle rows is done entirely inside one
// lane's 8 accumulator VGPRs (WMMA C layout: M = i + 8*(lane>=16), N = lane&15).
// ---------------------------------------------------------------------------

#define NLOC 2048
#define NNEI 32
#define ASEL 8
#define N_EDGE (NLOC * NNEI)     // 65536
#define N_ANGLE (N_EDGE * ASEL)  // 524288

#define INV_SQRT_DYNA 1.1180339887498949f  // 1/sqrt(0.8)
#define INV_DYNE 0.3125f                   // 1/3.2

typedef __attribute__((ext_vector_type(16))) __bf16 v16bf;
typedef __attribute__((ext_vector_type(8))) float v8f;
typedef unsigned short u16;

// packed-weight offsets (in bf16 elements) inside workspace
constexpr int OFF_LAEM_WG = 0;
constexpr int OFF_LAEM_WU = 20480;
constexpr int OFF_LAEM_WO = 40960;
constexpr int OFF_LAA_W   = 49152;
constexpr int OFF_LAW     = 54272;
constexpr int OFF_AAEM_WG = 56320;
constexpr int OFF_AAEM_WU = 97280;
constexpr int OFF_AAEM_WO = 138240;
constexpr int OFF_AAW     = 146432;
constexpr int OFF_AANM_WG = 150528;
constexpr int OFF_AANM_WU = 175104;
constexpr int OFF_AANM_WO = 199680;
constexpr int OFF_LRM_WG  = 216064;
constexpr int OFF_LRM_WU  = 252928;
constexpr int OFF_LRM_WO  = 289792;
constexpr int OFF_LRE     = 297984;
constexpr int OFF_LRN_W   = 300032;
constexpr int OFF_LREDGE_W= 304128;
constexpr int OFF_ARM_WG  = 306176;
constexpr int OFF_ARM_WU  = 347136;
constexpr int OFF_ARM_WO  = 388096;
constexpr int OFF_ARE     = 396288;
constexpr int OFF_ARN_W   = 398336;
constexpr int OFF_AREDGE_W= 406528;

static __device__ __forceinline__ u16 f2bf(float f) {
  union { float f; unsigned u; } v; v.f = f;
  return (u16)((v.u + 0x7FFFu + ((v.u >> 16) & 1u)) >> 16);
}
static __device__ __forceinline__ float siluf(float x) { return x / (1.f + __expf(-x)); }
static __device__ __forceinline__ float sigmf(float x) { return 1.f / (1.f + __expf(-x)); }

static __device__ __forceinline__ v16bf frag2(uint4 a, uint4 b) {
  union { uint4 u[2]; v16bf v; } x; x.u[0] = a; x.u[1] = b; return x.v;
}
// A fragment (16x32 bf16, row-major LDS, stride in halves):
// lane l holds row l&15; k-chunks [kb,kb+8) and [kb+16,kb+24), kb = 8*(l>>4)
static __device__ __forceinline__ v16bf load_a_frag(const u16* A, int stride, int kt, int lane) {
  int row = lane & 15;
  int kb = kt * 32 + ((lane >> 4) << 3);
  const uint4* p = (const uint4*)(A + row * stride + kb);
  const uint4* q = (const uint4*)(A + row * stride + kb + 16);
  return frag2(*p, *q);
}
// B fragment from pre-packed weights: lane's 16 halves contiguous (32B)
static __device__ __forceinline__ v16bf load_b_frag(const u16* P, int kt, int nt, int NT, int lane) {
  const uint4* p = (const uint4*)(P + (((kt * NT) + nt) * 32 + lane) * 16);
  return frag2(p[0], p[1]);
}
template <int KT>
static __device__ __forceinline__ v8f gemm_tile(const u16* A, int stride, const u16* P,
                                                int nt, int NT, int lane) {
  v8f c = {};
#pragma unroll
  for (int kt = 0; kt < KT; ++kt) {
    v16bf a = load_a_frag(A, stride, kt, lane);
    v16bf b = load_b_frag(P, kt, nt, NT, lane);
    c = __builtin_amdgcn_wmma_f32_16x16x32_bf16(false, a, false, b, (short)0, c, false, false);
  }
  return c;
}

// -------------------------- weight packing --------------------------------
// src f32 [K][N] -> bf16 fragments [kt][nt][lane][16], K zero-padded to 32.
// B element e of lane l: col = nt*16 + (l&15), k = kt*32 + 8*(l>>4) + (e&7) + 16*(e>>3)
__global__ void __launch_bounds__(32) k_pack(const float* w, u16* dst, int K, int N) {
  int NT = N >> 4;
  int tile = blockIdx.x;          // kt*NT + nt
  int nt = tile % NT;
  int lane = threadIdx.x;
  int col = nt * 16 + (lane & 15);
  int kb = (tile / NT) * 32 + ((lane >> 4) << 3);
  u16* o = dst + (tile * 32 + lane) * 16;
#pragma unroll
  for (int e = 0; e < 16; ++e) {
    int k = kb + (e & 7) + ((e >> 3) << 4);
    o[e] = f2bf((k < K) ? w[k * N + col] : 0.f);
  }
}

// ---------------- stage 1: line->edge messages + angle update --------------
// per block: 16 angle rows == 2 complete edges (ASEL=8)
__global__ void __launch_bounds__(256) k_line_msg(
    const float* angle_ebd, const float* edge_ebd, const float* a_sw, const int* eik2a,
    const u16* W, const float* bg, const float* bu, const float* bo,
    const float* laa_b, const float* res_lae, const float* res_laa,
    float* out_edge, float* out_angle) {
  __shared__ __align__(16) u16 ldsA[16 * 160];
  __shared__ __align__(16) u16 ldsH[16 * 128];
  __shared__ float ldsW[16 * 64];
  __shared__ float ldsASW[16];
  const int tid = threadIdx.x, lane = tid & 31, wave = tid >> 5;
  const int abase = blockIdx.x * 16;

  for (int idx = tid; idx < 16 * 160; idx += 256) {
    int r = idx / 160, c = idx - r * 160;
    int a = abase + r;
    float v;
    if (c < 32)      v = angle_ebd[a * 32 + c];
    else if (c < 96) v = edge_ebd[eik2a[a] * 64 + (c - 32)];
    else             v = edge_ebd[(a >> 3) * 64 + (c - 96)];
    ldsA[idx] = f2bf(v);
  }
  if (tid < 16) ldsASW[tid] = a_sw[abase + tid];
  __syncthreads();

  v8f g = gemm_tile<5>(ldsA, 160, W + OFF_LAEM_WG, wave, 8, lane);
  v8f u = gemm_tile<5>(ldsA, 160, W + OFF_LAEM_WU, wave, 8, lane);
  const int mb = (lane >> 4) << 3;
  {
    int col = wave * 16 + (lane & 15);
    float bgv = bg[col], buv = bu[col];
#pragma unroll
    for (int i = 0; i < 8; ++i)
      ldsH[(mb + i) * 128 + col] = f2bf(siluf(g[i] + bgv) * (u[i] + buv));
  }
  __syncthreads();

  v8f o = {};
  if (wave < 4) {
    o = gemm_tile<4>(ldsH, 128, W + OFF_LAEM_WO, wave, 4, lane);
  } else {
    int nt = wave - 4;
    v8f lg = gemm_tile<1>(ldsA, 160, W + OFF_LAW, nt, 4, lane);
    int col = nt * 16 + (lane & 15);
    float l[8], mx = -1e30f;
#pragma unroll
    for (int i = 0; i < 8; ++i) { l[i] = lg[i] * ldsASW[mb + i]; mx = fmaxf(mx, l[i]); }
    float s = 0.f, ex[8];
#pragma unroll
    for (int i = 0; i < 8; ++i) { ex[i] = __expf(l[i] - mx); s += ex[i]; }
    float inv = 1.f / s;
#pragma unroll
    for (int i = 0; i < 8; ++i) ldsW[(mb + i) * 64 + col] = ex[i] * inv * ldsASW[mb + i];
  }
  __syncthreads();

  if (wave < 4) {  // edge aggregation: one edge per half-wave, intra-lane sum
    int col = wave * 16 + (lane & 15);
    float bov = bo[col], agg = 0.f;
#pragma unroll
    for (int i = 0; i < 8; ++i) agg += ldsW[(mb + i) * 64 + col] * (o[i] + bov);
    int e = (abase >> 3) + (lane >> 4);
    out_edge[e * 64 + col] = agg * INV_SQRT_DYNA + res_lae[col] * edge_ebd[e * 64 + col];
  } else if (wave < 6) {  // angle update (out 32 -> 2 n-tiles)
    int nt = wave - 4;
    v8f aa = gemm_tile<5>(ldsA, 160, W + OFF_LAA_W, nt, 2, lane);
    int col = nt * 16 + (lane & 15);
    float b = laa_b[col], rl = res_laa[col];
#pragma unroll
    for (int i = 0; i < 8; ++i) {
      int r = abase + mb + i;
      out_angle[r * 32 + col] = siluf(aa[i] + b) + rl * angle_ebd[r * 32 + col];
    }
  }
}

// --------- stage 2A: edge->node attention messages (one node/block) --------
__global__ void __launch_bounds__(256) k_atom_msg(
    const float* node_ext, const float* edge1, const float* sw, const int* ne2e,
    const u16* W, const float* bg, const float* bu, const float* bo,
    const float* res_aae, float* edge2, float* nodeAgg) {
  __shared__ __align__(16) u16 ldsA[32 * 320];
  __shared__ __align__(16) u16 ldsH[32 * 128];
  __shared__ float ldsL[32 * 64];
  __shared__ float ldsMx[64], ldsInv[64], aggBuf[64];
  __shared__ float ldsSW[32];
  const int tid = threadIdx.x, lane = tid & 31, wave = tid >> 5;
  const int n = blockIdx.x, ebase = n * 32;

  for (int idx = tid; idx < 32 * 320; idx += 256) {
    int r = idx / 320, c = idx - r * 320;
    int e = ebase + r;
    float v;
    if (c < 128)      v = node_ext[n * 128 + c];
    else if (c < 256) v = node_ext[ne2e[e] * 128 + (c - 128)];
    else              v = edge1[e * 64 + (c - 256)];
    ldsA[idx] = f2bf(v);
  }
  if (tid < 32) ldsSW[tid] = sw[ebase + tid];
  if (tid < 64) aggBuf[tid] = 0.f;
  __syncthreads();

#pragma unroll
  for (int m = 0; m < 2; ++m) {
    v8f g = gemm_tile<10>(ldsA + m * 16 * 320, 320, W + OFF_AAEM_WG, wave, 8, lane);
    v8f u = gemm_tile<10>(ldsA + m * 16 * 320, 320, W + OFF_AAEM_WU, wave, 8, lane);
    int col = wave * 16 + (lane & 15);
    float bgv = bg[col], buv = bu[col];
    int mb2 = (lane >> 4) << 3;
#pragma unroll
    for (int i = 0; i < 8; ++i)
      ldsH[(m * 16 + mb2 + i) * 128 + col] = f2bf(siluf(g[i] + bgv) * (u[i] + buv));
  }
  __syncthreads();

  const int mt = wave >> 2, nt = wave & 3;
  const int col = nt * 16 + (lane & 15);
  const int row0 = mt * 16 + ((lane >> 4) << 3);
  v8f o  = gemm_tile<4>(ldsH + mt * 16 * 128, 128, W + OFF_AAEM_WO, nt, 4, lane);
  v8f lg = gemm_tile<2>(ldsA + mt * 16 * 320 + 256, 320, W + OFF_AAW, nt, 4, lane);
#pragma unroll
  for (int i = 0; i < 8; ++i) ldsL[(row0 + i) * 64 + col] = lg[i] * ldsSW[row0 + i];
  __syncthreads();
  if (tid < 64) {  // per-column softmax stats over 32 rows
    float mx = -1e30f;
    for (int r = 0; r < 32; ++r) mx = fmaxf(mx, ldsL[r * 64 + tid]);
    float s = 0.f;
    for (int r = 0; r < 32; ++r) s += __expf(ldsL[r * 64 + tid] - mx);
    ldsMx[tid] = mx; ldsInv[tid] = 1.f / s;
  }
  __syncthreads();
  {
    float bov = bo[col], ra = res_aae[col];
    float mx = ldsMx[col], inv = ldsInv[col], agg = 0.f;
#pragma unroll
    for (int i = 0; i < 8; ++i) {
      int r = row0 + i, e = ebase + r;
      float up = o[i] + bov;
      agg += __expf(ldsL[r * 64 + col] - mx) * inv * ldsSW[r] * up;
      edge2[e * 64 + col] = up + ra * edge1[e * 64 + col];
    }
    atomicAdd(&aggBuf[col], agg);
  }
  __syncthreads();
  if (tid < 64) nodeAgg[n * 64 + tid] = aggBuf[tid] * INV_DYNE;
}

// ----------------------- stage 2B: node update -----------------------------
__global__ void __launch_bounds__(256) k_node_update(
    const float* node_ext, const float* nodeAgg, const u16* W,
    const float* bg, const float* bu, const float* bo, const float* res_aan,
    float* out_node) {
  __shared__ __align__(16) u16 ldsA[16 * 192];
  __shared__ __align__(16) u16 ldsH[16 * 128];
  const int tid = threadIdx.x, lane = tid & 31, wave = tid >> 5;
  const int nbase = blockIdx.x * 16;
  for (int idx = tid; idx < 16 * 192; idx += 256) {
    int r = idx / 192, c = idx - r * 192;
    float v = (c < 128) ? node_ext[(nbase + r) * 128 + c] : nodeAgg[(nbase + r) * 64 + (c - 128)];
    ldsA[idx] = f2bf(v);
  }
  __syncthreads();
  v8f g = gemm_tile<6>(ldsA, 192, W + OFF_AANM_WG, wave, 8, lane);
  v8f u = gemm_tile<6>(ldsA, 192, W + OFF_AANM_WU, wave, 8, lane);
  const int col = wave * 16 + (lane & 15);
  const int mb = (lane >> 4) << 3;
  {
    float bgv = bg[col], buv = bu[col];
#pragma unroll
    for (int i = 0; i < 8; ++i)
      ldsH[(mb + i) * 128 + col] = f2bf(siluf(g[i] + bgv) * (u[i] + buv));
  }
  __syncthreads();
  v8f o = gemm_tile<4>(ldsH, 128, W + OFF_AANM_WO, wave, 8, lane);
  float bov = bo[col], rn = res_aan[col];
#pragma unroll
  for (int i = 0; i < 8; ++i) {
    int r = nbase + mb + i;
    out_node[r * 128 + col] = (o[i] + bov) + rn * node_ext[r * 128 + col];
  }
}

// --------------------- stage 3A: line refinement ---------------------------
__global__ void __launch_bounds__(256) k_line_refine(
    const float* angle1, const float* node2, const float* edge2,
    const float* a_sw, const float* angle_rbf, const int* eik2a, const u16* W,
    const float* bg, const float* bu, const float* bo,
    const float* lredge_b, const float* res_lra,
    float* lineAgg, float* out_angle /* aliases angle1, rows owned by block */) {
  __shared__ __align__(16) u16 ldsA[16 * 288];
  __shared__ __align__(16) u16 ldsH[16 * 128];
  __shared__ __align__(16) u16 ldsR[16 * 32];
  __shared__ __align__(16) u16 ldsF[16 * 64];
  __shared__ float ldsASW[16];
  const int tid = threadIdx.x, lane = tid & 31, wave = tid >> 5;
  const int abase = blockIdx.x * 16;
  for (int idx = tid; idx < 16 * 288; idx += 256) {
    int r = idx / 288, c = idx - r * 288;
    int a = abase + r;
    float v;
    if (c < 32)       v = angle1[a * 32 + c];
    else if (c < 160) v = node2[(a >> 8) * 128 + (c - 32)];   // n2a = a>>8
    else if (c < 224) v = edge2[eik2a[a] * 64 + (c - 160)];
    else              v = edge2[(a >> 3) * 64 + (c - 224)];   // eij2a = a>>3
    ldsA[idx] = f2bf(v);
  }
  for (int idx = tid; idx < 16 * 32; idx += 256) {
    int r = idx >> 5, c = idx & 31;
    ldsR[idx] = f2bf((c < 7) ? angle_rbf[(abase + r) * 7 + c] : 0.f);
  }
  if (tid < 16) ldsASW[tid] = a_sw[abase + tid];
  __syncthreads();
  v8f g = gemm_tile<9>(ldsA, 288, W + OFF_LRM_WG, wave, 8, lane);
  v8f u = gemm_tile<9>(ldsA, 288, W + OFF_LRM_WU, wave, 8, lane);
  const int mb = (lane >> 4) << 3;
  {
    int col = wave * 16 + (lane & 15);
    float bgv = bg[col], buv = bu[col];
#pragma unroll
    for (int i = 0; i < 8; ++i)
      ldsH[(mb + i) * 128 + col] = f2bf(siluf(g[i] + bgv) * (u[i] + buv));
  }
  __syncthreads();
  if (wave < 4) {
    v8f o  = gemm_tile<4>(ldsH, 128, W + OFF_LRM_WO, wave, 4, lane);
    v8f ev = gemm_tile<1>(ldsR, 32, W + OFF_LRE, wave, 4, lane);
    int col = wave * 16 + (lane & 15);
    float bov = bo[col], agg = 0.f;
#pragma unroll
    for (int i = 0; i < 8; ++i) {
      float f = o[i] + bov;
      ldsF[(mb + i) * 64 + col] = f2bf(f);
      agg += f * sigmf(ev[i]) * ldsASW[mb + i];
    }
    int e = (abase >> 3) + (lane >> 4);
    lineAgg[e * 64 + col] = agg * INV_SQRT_DYNA;
  }
  __syncthreads();
  if (wave < 2) {  // delta_angle = silu(feat @ lredge + b) + res_lra*angle1
    v8f d = gemm_tile<2>(ldsF, 64, W + OFF_LREDGE_W, wave, 2, lane);
    int col = wave * 16 + (lane & 15);
    float b = lredge_b[col], rl = res_lra[col];
#pragma unroll
    for (int i = 0; i < 8; ++i) {
      int r = abase + mb + i;
      out_angle[r * 32 + col] = siluf(d[i] + b) + rl * angle1[r * 32 + col];
    }
  }
}

// --------------------- stage 3B: edge refinement ---------------------------
__global__ void __launch_bounds__(128) k_edge_refine(
    const float* lineAgg, const float* edge2, const u16* W,
    const float* lrn_b, const float* res_lre, float* out_edge) {
  __shared__ __align__(16) u16 ldsA[16 * 64];
  const int tid = threadIdx.x, lane = tid & 31, wave = tid >> 5;
  const int ebase = blockIdx.x * 16;
  for (int idx = tid; idx < 16 * 64; idx += 128) ldsA[idx] = f2bf(lineAgg[ebase * 64 + idx]);
  __syncthreads();
  v8f o = gemm_tile<2>(ldsA, 64, W + OFF_LRN_W, wave, 4, lane);
  int col = wave * 16 + (lane & 15);
  int mb = (lane >> 4) << 3;
  float b = lrn_b[col], rl = res_lre[col];
#pragma unroll
  for (int i = 0; i < 8; ++i) {
    int e = ebase + mb + i;
    out_edge[e * 64 + col] = siluf(o[i] + b) + rl * edge2[e * 64 + col];
  }
}

// --------------------- stage 4A: atom refinement ---------------------------
__global__ void __launch_bounds__(256) k_atom_refine(
    const float* node2, const float* node_ext, const float* edge3,
    const float* sw, const float* edge_rbf, const int* ne2e, const u16* W,
    const float* bg, const float* bu, const float* bo,
    const float* aredge_b, const float* res_are,
    float* atomAgg, float* out_edge /* aliases edge3, rows owned by block */) {
  __shared__ __align__(16) u16 ldsA[32 * 320];
  __shared__ __align__(16) u16 ldsH[32 * 128];
  __shared__ __align__(16) u16 ldsR[32 * 32];
  __shared__ __align__(16) u16 ldsF[32 * 64];
  __shared__ float aggBuf[64];
  __shared__ float ldsSW[32];
  const int tid = threadIdx.x, lane = tid & 31, wave = tid >> 5;
  const int n = blockIdx.x, ebase = n * 32;
  for (int idx = tid; idx < 32 * 320; idx += 256) {
    int r = idx / 320, c = idx - r * 320;
    int e = ebase + r;
    float v;
    if (c < 128)      v = node2[n * 128 + c];
    else if (c < 256) v = node_ext[ne2e[e] * 128 + (c - 128)];
    else              v = edge3[e * 64 + (c - 256)];
    ldsA[idx] = f2bf(v);
  }
  for (int idx = tid; idx < 32 * 32; idx += 256) {
    int r = idx >> 5, c = idx & 31;
    ldsR[idx] = f2bf((c < 7) ? edge_rbf[(ebase + r) * 7 + c] : 0.f);
  }
  if (tid < 32) ldsSW[tid] = sw[ebase + tid];
  if (tid < 64) aggBuf[tid] = 0.f;
  __syncthreads();
#pragma unroll
  for (int m = 0; m < 2; ++m) {
    v8f g = gemm_tile<10>(ldsA + m * 16 * 320, 320, W + OFF_ARM_WG, wave, 8, lane);
    v8f u = gemm_tile<10>(ldsA + m * 16 * 320, 320, W + OFF_ARM_WU, wave, 8, lane);
    int col = wave * 16 + (lane & 15);
    float bgv = bg[col], buv = bu[col];
    int mb2 = (lane >> 4) << 3;
#pragma unroll
    for (int i = 0; i < 8; ++i)
      ldsH[(m * 16 + mb2 + i) * 128 + col] = f2bf(siluf(g[i] + bgv) * (u[i] + buv));
  }
  __syncthreads();
  const int mt = wave >> 2, nt = wave & 3;
  const int col = nt * 16 + (lane & 15);
  const int row0 = mt * 16 + ((lane >> 4) << 3);
  {
    v8f o  = gemm_tile<4>(ldsH + mt * 16 * 128, 128, W + OFF_ARM_WO, nt, 4, lane);
    v8f ev = gemm_tile<1>(ldsR + mt * 16 * 32, 32, W + OFF_ARE, nt, 4, lane);
    float bov = bo[col], agg = 0.f;
#pragma unroll
    for (int i = 0; i < 8; ++i) {
      float f = o[i] + bov;
      ldsF[(row0 + i) * 64 + col] = f2bf(f);
      agg += f * sigmf(ev[i]) * ldsSW[row0 + i];
    }
    atomicAdd(&aggBuf[col], agg);
  }
  __syncthreads();
  if (tid < 64) atomAgg[n * 64 + tid] = aggBuf[tid] * INV_DYNE;
  {
    v8f d = gemm_tile<2>(ldsF + mt * 16 * 64, 64, W + OFF_AREDGE_W, nt, 4, lane);
    float b = aredge_b[col], ra = res_are[col];
#pragma unroll
    for (int i = 0; i < 8; ++i) {
      int e = ebase + row0 + i;
      out_edge[e * 64 + col] = siluf(d[i] + b) + ra * edge3[e * 64 + col];
    }
  }
}

// --------------------- stage 4B: node refinement ---------------------------
__global__ void __launch_bounds__(256) k_node_refine(
    const float* atomAgg, const u16* W, const float* arn_b, const float* res_arn,
    float* out_node /* holds node2, overwritten in place */) {
  __shared__ __align__(16) u16 ldsA[16 * 64];
  const int tid = threadIdx.x, lane = tid & 31, wave = tid >> 5;
  const int nbase = blockIdx.x * 16;
  for (int idx = tid; idx < 16 * 64; idx += 256) ldsA[idx] = f2bf(atomAgg[nbase * 64 + idx]);
  __syncthreads();
  v8f o = gemm_tile<2>(ldsA, 64, W + OFF_ARN_W, wave, 8, lane);
  int col = wave * 16 + (lane & 15);
  int mb = (lane >> 4) << 3;
  float b = arn_b[col], rn = res_arn[col];
#pragma unroll
  for (int i = 0; i < 8; ++i) {
    int r = nbase + mb + i;
    out_node[r * 128 + col] = siluf(o[i] + b) + rn * out_node[r * 128 + col];
  }
}

// ---------------------------------------------------------------------------
extern "C" void kernel_launch(void* const* d_in, const int* in_sizes, int n_in,
                              void* d_out, int out_size, void* d_ws, size_t ws_size,
                              hipStream_t stream) {
  (void)in_sizes; (void)n_in; (void)out_size; (void)ws_size;
  // positional inputs (setup_inputs insertion order), then params leaves in
  // jax.tree_util sorted-key order (gmlp: bg,bo,bu,wg,wo,wu ; mlp: b,w)
  const float* node_ext  = (const float*)d_in[0];
  const float* edge_ebd  = (const float*)d_in[1];
  const float* angle_ebd = (const float*)d_in[3];
  const float* sw        = (const float*)d_in[5];
  const float* a_sw      = (const float*)d_in[6];
  const int*   edge_index  = (const int*)d_in[7];
  const int*   angle_index = (const int*)d_in[8];
  const float* edge_rbf  = (const float*)d_in[9];
  const float* angle_rbf = (const float*)d_in[10];
  const int* ne2e  = edge_index + N_EDGE;
  const int* eik2a = angle_index + 2 * N_ANGLE;

  u16* W = (u16*)d_ws;                                   // 1MB packed weights
  float* fws     = (float*)((char*)d_ws + (1 << 20));
  float* edge2   = fws;                                  // 65536*64
  float* lineAgg = edge2 + (size_t)N_EDGE * 64;          // 65536*64
  float* nodeAgg = lineAgg + (size_t)N_EDGE * 64;        // 2048*64
  float* atomAgg = nodeAgg + (size_t)NLOC * 64;          // 2048*64

  float* out_node  = (float*)d_out;                      // 2048*128
  float* out_edge  = out_node + (size_t)NLOC * 128;      // 65536*64 (edge1/edge3/e_upd)
  float* out_angle = out_edge + (size_t)N_EDGE * 64;     // 524288*32 (angle1/final)

  struct Job { int idx, K, N, off; };
  const Job jobs[] = {
    {40,160,128,OFF_LAEM_WG}, {42,160,128,OFF_LAEM_WU}, {41,128, 64,OFF_LAEM_WO},
    {36,160, 32,OFF_LAA_W},   {43, 32, 64,OFF_LAW},
    {14,320,128,OFF_AAEM_WG}, {16,320,128,OFF_AAEM_WU}, {15,128, 64,OFF_AAEM_WO},
    {23, 64, 64,OFF_AAW},
    {20,192,128,OFF_AANM_WG}, {22,192,128,OFF_AANM_WU}, {21,128,128,OFF_AANM_WO},
    {50,288,128,OFF_LRM_WG},  {52,288,128,OFF_LRM_WU},  {51,128, 64,OFF_LRM_WO},
    {44,  7, 64,OFF_LRE},     {54, 64, 64,OFF_LRN_W},   {46, 64, 32,OFF_LREDGE_W},
    {30,320,128,OFF_ARM_WG},  {32,320,128,OFF_ARM_WU},  {31,128, 64,OFF_ARM_WO},
    {24,  7, 64,OFF_ARE},     {34, 64,128,OFF_ARN_W},   {26, 64, 64,OFF_AREDGE_W},
  };
  for (const Job& j : jobs) {
    int kt = (j.K + 31) / 32;
    k_pack<<<kt * (j.N / 16), 32, 0, stream>>>((const float*)d_in[j.idx], W + j.off, j.K, j.N);
  }

  k_line_msg<<<N_ANGLE / 16, 256, 0, stream>>>(
      angle_ebd, edge_ebd, a_sw, eik2a, W,
      (const float*)d_in[37], (const float*)d_in[39], (const float*)d_in[38],  // laem bg,bu,bo
      (const float*)d_in[35], (const float*)d_in[59], (const float*)d_in[60],  // laa.b,res_lae,res_laa
      out_edge, out_angle);

  k_atom_msg<<<NLOC, 256, 0, stream>>>(
      node_ext, out_edge, sw, ne2e, W,
      (const float*)d_in[11], (const float*)d_in[13], (const float*)d_in[12],  // aaem bg,bu,bo
      (const float*)d_in[55], edge2, nodeAgg);                                 // res_aae

  k_node_update<<<NLOC / 16, 256, 0, stream>>>(
      node_ext, nodeAgg, W,
      (const float*)d_in[17], (const float*)d_in[19], (const float*)d_in[18],  // aanm bg,bu,bo
      (const float*)d_in[56], out_node);                                       // res_aan

  k_line_refine<<<N_ANGLE / 16, 256, 0, stream>>>(
      out_angle, out_node, edge2, a_sw, angle_rbf, eik2a, W,
      (const float*)d_in[47], (const float*)d_in[49], (const float*)d_in[48],  // lrm bg,bu,bo
      (const float*)d_in[45], (const float*)d_in[61],                          // lredge.b,res_lra
      lineAgg, out_angle);

  k_edge_refine<<<N_EDGE / 16, 128, 0, stream>>>(
      lineAgg, edge2, W,
      (const float*)d_in[53], (const float*)d_in[62], out_edge);               // lrn.b,res_lre

  k_atom_refine<<<NLOC, 256, 0, stream>>>(
      out_node, node_ext, out_edge, sw, edge_rbf, ne2e, W,
      (const float*)d_in[27], (const float*)d_in[29], (const float*)d_in[28],  // arm bg,bu,bo
      (const float*)d_in[25], (const float*)d_in[57],                          // aredge.b,res_are
      atomAgg, out_edge);

  k_node_refine<<<NLOC / 16, 256, 0, stream>>>(
      atomAgg, W,
      (const float*)d_in[33], (const float*)d_in[58], out_node);               // arn.b,res_arn
}
